// BertLayer_89309549953365
// MI455X (gfx1250) — matrix operand order repaired and checked by reading, not compiled, not run on hardware
//
#include <hip/hip_runtime.h>
#include <hip/hip_bf16.h>

// BERT layer: B=8, S=1024, E=1024, F=4096, fp32 I/O, bf16 WMMA internals.
#define BB 8
#define SS 1024
#define EE 1024
#define FFD 4096

typedef __bf16 bf16_t;
typedef __attribute__((ext_vector_type(8)))  bf16_t v8bf;
typedef __attribute__((ext_vector_type(16))) bf16_t v16bf;
typedef __attribute__((ext_vector_type(8)))  float  v8f;

__device__ __forceinline__ bf16_t f2bf(float f) {
  union { float f; unsigned u; } c; c.f = f;
  unsigned r = c.u + 0x7FFFu + ((c.u >> 16) & 1u);   // round-to-nearest-even
  union { unsigned short s; bf16_t b; } o; o.s = (unsigned short)(r >> 16);
  return o.b;
}

// ---------------------------------------------------------------------------
// fp32 -> bf16 elementwise
// ---------------------------------------------------------------------------
__global__ __launch_bounds__(256)
void cvt_f32_bf16_kernel(const float* __restrict__ in, bf16_t* __restrict__ out, long n) {
  long i = (long)blockIdx.x * blockDim.x + threadIdx.x;
  long stride = (long)gridDim.x * blockDim.x;
  for (; i < n; i += stride) out[i] = f2bf(in[i]);
}

// fp32 [R][C] -> bf16 [C][R] (transpose + convert), 32x32 tiles
__global__ __launch_bounds__(256)
void transpose_cvt_kernel(const float* __restrict__ in, bf16_t* __restrict__ out,
                          int R, int C) {
  __shared__ float tile[32][33];
  const int bx = blockIdx.x * 32;   // along C
  const int by = blockIdx.y * 32;   // along R
  const int tx = threadIdx.x;       // 0..31
  const int ty = threadIdx.y;       // 0..7
#pragma unroll
  for (int j = 0; j < 4; ++j)
    tile[ty + j * 8][tx] = in[(size_t)(by + ty + j * 8) * C + bx + tx];
  __syncthreads();
#pragma unroll
  for (int j = 0; j < 4; ++j)
    out[(size_t)(bx + ty + j * 8) * R + by + tx] = f2bf(tile[tx][ty + j * 8]);
}

// ---------------------------------------------------------------------------
// bf16 WMMA GEMM: C[M,N] = A[M,K] * Bt[N,K]^T  (+bias +residual, act, stores)
// 128x128 block tile, BK=64, 256 threads = 8 waves, wave tile 64x32 (4x2 WMMA)
// Double-buffered LDS, register-staged b128 global loads.
// ---------------------------------------------------------------------------
template<bool HAS_BIAS, bool HAS_RES, bool GELU_ACT,
         bool STORE_F32, bool STORE_BF16, bool TRANS_BF16>
__global__ __launch_bounds__(256)
void gemm_bf16_kernel(const bf16_t* __restrict__ A, const bf16_t* __restrict__ Bt,
                      const float* __restrict__ bias, const float* __restrict__ res,
                      float* __restrict__ Cf, bf16_t* __restrict__ Cb,
                      int M, int N, int K,
                      long aBatch, long bBatch, long cfBatch, long cbBatch,
                      int transRPB, int transLd, long transBS) {
  const int z = blockIdx.z;
  A  += (long)z * aBatch;
  Bt += (long)z * bBatch;
  if (STORE_F32)  Cf += (long)z * cfBatch;
  if (STORE_BF16) Cb += (long)z * cbBatch;
  if (HAS_RES)    res += (long)z * cfBatch;

  constexpr int BK = 64, LDT = 72;   // LDS row stride (pad: 144B, 16B-aligned)
  __shared__ bf16_t As[2][128 * LDT];
  __shared__ bf16_t Bs[2][128 * LDT];

  const int tid  = threadIdx.x;
  const int m0   = blockIdx.y * 128;
  const int n0   = blockIdx.x * 128;
  const int w    = tid >> 5, lane = tid & 31;
  const int wm   = (w >> 2) * 64;    // wave M offset: 0/64
  const int wn   = (w & 3) * 32;     // wave N offset: 0/32/64/96
  const int hi   = lane >> 4;        // half-wave select
  const int l    = lane & 15;

  v8f acc[4][2];
#pragma unroll
  for (int i = 0; i < 4; ++i)
#pragma unroll
    for (int j = 0; j < 2; ++j) acc[i][j] = v8f{};

  // global staging: 128x64 tile, b128 loads: row = tid>>3, col = (tid&7)*8
  const int gr = tid >> 3;           // 0..31
  const int gc = (tid & 7) << 3;     // 0..56 step 8
  v8bf aReg[4], bReg[4];

  auto loadRegs = [&](int k0) {
#pragma unroll
    for (int i = 0; i < 4; ++i) {
      int r = gr + i * 32;
      aReg[i] = *(const v8bf*)&A [(size_t)(m0 + r) * K + k0 + gc];
      bReg[i] = *(const v8bf*)&Bt[(size_t)(n0 + r) * K + k0 + gc];
    }
  };
  auto storeLDS = [&](int buf) {
#pragma unroll
    for (int i = 0; i < 4; ++i) {
      int r = gr + i * 32;
      *(v8bf*)&As[buf][r * LDT + gc] = aReg[i];
      *(v8bf*)&Bs[buf][r * LDT + gc] = bReg[i];
    }
  };

  const int nT = K / BK;
  loadRegs(0);
  storeLDS(0);
  __syncthreads();

  for (int t = 0; t < nT; ++t) {
    if (t + 1 < nT) loadRegs((t + 1) * BK);   // overlap with compute below

    const int buf = t & 1;
#pragma unroll
    for (int ks = 0; ks < 2; ++ks) {
      v16bf af[4], bfm[2];
#pragma unroll
      for (int mt = 0; mt < 4; ++mt) {
        // A-matrix 16x32 layout: lane<16: M=l, K {0..7,16..23}; lane>=16: K {8..15,24..31}
        int row = wm + mt * 16 + l;
        v8bf lo = *(const v8bf*)&As[buf][row * LDT + ks * 32 + hi * 8];
        v8bf hc = *(const v8bf*)&As[buf][row * LDT + ks * 32 + 16 + hi * 8];
        af[mt] = __builtin_shufflevector(lo, hc, 0,1,2,3,4,5,6,7,8,9,10,11,12,13,14,15);
      }
#pragma unroll
      for (int nt = 0; nt < 2; ++nt) {
        // B-matrix 32x16: lane<16: N=l, K {0..7,16..23}; lane>=16: K {8..15,24..31}
        int col = wn + nt * 16 + l;
        v8bf lo = *(const v8bf*)&Bs[buf][col * LDT + ks * 32 + hi * 8];
        v8bf hc = *(const v8bf*)&Bs[buf][col * LDT + ks * 32 + 16 + hi * 8];
        bfm[nt] = __builtin_shufflevector(lo, hc, 0,1,2,3,4,5,6,7,8,9,10,11,12,13,14,15);
      }
#pragma unroll
      for (int mt = 0; mt < 4; ++mt)
#pragma unroll
        for (int nt = 0; nt < 2; ++nt)
          acc[mt][nt] = __builtin_amdgcn_wmma_f32_16x16x32_bf16(
              false, af[mt], false, bfm[nt], (short)0, acc[mt][nt], false, false);
    }

    if (t + 1 < nT) storeLDS((t + 1) & 1);    // other buffer, last read at t-1
    __syncthreads();
  }

  // epilogue: C layout — VGPR r: M = r + 8*hi, N = l (within 16x16 tile)
#pragma unroll
  for (int mt = 0; mt < 4; ++mt) {
#pragma unroll
    for (int nt = 0; nt < 2; ++nt) {
      const int colg = n0 + wn + nt * 16 + l;
      const int rowb = m0 + wm + mt * 16 + hi * 8;
#pragma unroll
      for (int r = 0; r < 8; ++r) {
        const int row = rowb + r;
        float v = acc[mt][nt][r];
        if (HAS_BIAS) v += bias[colg];
        if (HAS_RES)  v += res[(size_t)row * N + colg];
        if (GELU_ACT) v = 0.5f * v * (1.0f + erff(v * 0.70710678118654752f));
        if (STORE_F32) Cf[(size_t)row * N + colg] = v;
        if (STORE_BF16) {
          if (TRANS_BF16) {
            int bz = row / transRPB;
            int rm = row - bz * transRPB;
            Cb[(long)bz * transBS + (long)colg * transLd + rm] = f2bf(v);
          } else {
            Cb[(size_t)row * N + colg] = f2bf(v);
          }
        }
      }
    }
  }
}

// ---------------------------------------------------------------------------
// row softmax over 1024 cols (scale folded in), fp32 in -> bf16 out
// ---------------------------------------------------------------------------
__global__ __launch_bounds__(256)
void softmax_bf16_kernel(const float* __restrict__ X, bf16_t* __restrict__ Y,
                         float scale) {
  const int row = blockIdx.x;
  const float* x = X + (size_t)row * SS;
  bf16_t* y = Y + (size_t)row * SS;
  const int tid = threadIdx.x;
  const int w = tid >> 5, lane = tid & 31;
  __shared__ float redm[8];
  __shared__ float reds[8];

  float t[4];
  float m = -3.4e38f;
#pragma unroll
  for (int i = 0; i < 4; ++i) { t[i] = x[tid + i * 256] * scale; m = fmaxf(m, t[i]); }
#pragma unroll
  for (int off = 16; off > 0; off >>= 1) m = fmaxf(m, __shfl_xor(m, off, 32));
  if (lane == 0) redm[w] = m;
  __syncthreads();
  float bm = redm[0];
#pragma unroll
  for (int i = 1; i < 8; ++i) bm = fmaxf(bm, redm[i]);

  float s = 0.f;
#pragma unroll
  for (int i = 0; i < 4; ++i) { t[i] = __expf(t[i] - bm); s += t[i]; }
#pragma unroll
  for (int off = 16; off > 0; off >>= 1) s += __shfl_xor(s, off, 32);
  if (lane == 0) reds[w] = s;
  __syncthreads();
  float bs = 0.f;
#pragma unroll
  for (int i = 0; i < 8; ++i) bs += reds[i];
  const float inv = 1.0f / bs;
#pragma unroll
  for (int i = 0; i < 4; ++i) y[tid + i * 256] = f2bf(t[i] * inv);
}

// ---------------------------------------------------------------------------
// row layernorm over 1024 cols, fp32 in -> fp32 out (+optional bf16 copy)
// ---------------------------------------------------------------------------
template<bool STORE_BF16>
__global__ __launch_bounds__(256)
void layernorm_kernel(const float* __restrict__ X, const float* __restrict__ g,
                      const float* __restrict__ b, float* __restrict__ Yf,
                      bf16_t* __restrict__ Yb) {
  const int row = blockIdx.x;
  const float* x = X + (size_t)row * EE;
  const int tid = threadIdx.x;
  const int w = tid >> 5, lane = tid & 31;
  __shared__ float red1[8];
  __shared__ float red2[8];

  float t[4], s = 0.f, s2 = 0.f;
#pragma unroll
  for (int i = 0; i < 4; ++i) { t[i] = x[tid + i * 256]; s += t[i]; s2 += t[i] * t[i]; }
#pragma unroll
  for (int off = 16; off > 0; off >>= 1) { s += __shfl_xor(s, off, 32); s2 += __shfl_xor(s2, off, 32); }
  if (lane == 0) { red1[w] = s; red2[w] = s2; }
  __syncthreads();
  float S = 0.f, S2 = 0.f;
#pragma unroll
  for (int i = 0; i < 8; ++i) { S += red1[i]; S2 += red2[i]; }
  const float mean = S * (1.0f / EE);
  const float var  = S2 * (1.0f / EE) - mean * mean;
  const float rstd = rsqrtf(var + 1e-12f);
#pragma unroll
  for (int i = 0; i < 4; ++i) {
    const int c = tid + i * 256;
    const float v = (t[i] - mean) * rstd * g[c] + b[c];
    Yf[(size_t)row * EE + c] = v;
    if (STORE_BF16) Yb[(size_t)row * EE + c] = f2bf(v);
  }
}

// ---------------------------------------------------------------------------
extern "C" void kernel_launch(void* const* d_in, const int* in_sizes, int n_in,
                              void* d_out, int out_size, void* d_ws, size_t ws_size,
                              hipStream_t stream) {
  const float* x  = (const float*)d_in[0];
  const float* Wq = (const float*)d_in[1];
  const float* bq = (const float*)d_in[2];
  const float* Wk = (const float*)d_in[3];
  const float* bk = (const float*)d_in[4];
  const float* Wv = (const float*)d_in[5];
  const float* bv = (const float*)d_in[6];
  const float* Wd = (const float*)d_in[7];
  const float* bd = (const float*)d_in[8];
  const float* g1 = (const float*)d_in[9];
  const float* b1 = (const float*)d_in[10];
  const float* Wi = (const float*)d_in[11];
  const float* bi = (const float*)d_in[12];
  const float* Wo = (const float*)d_in[13];
  const float* bo = (const float*)d_in[14];
  const float* g2 = (const float*)d_in[15];
  const float* b2 = (const float*)d_in[16];

  char* ws = (char*)d_ws;
  size_t off = 0;
  auto alloc = [&](size_t bytes) -> void* {
    void* p = ws + off;
    off = (off + bytes + 255) & ~(size_t)255;
    return p;
  };

  const long T = (long)BB * SS;  // 8192 token rows
  bf16_t* xb    = (bf16_t*)alloc(T * EE * 2);
  bf16_t* wqT   = (bf16_t*)alloc((long)EE * EE * 2);
  bf16_t* wkT   = (bf16_t*)alloc((long)EE * EE * 2);
  bf16_t* wvT   = (bf16_t*)alloc((long)EE * EE * 2);
  bf16_t* wdT   = (bf16_t*)alloc((long)EE * EE * 2);
  bf16_t* wiT   = (bf16_t*)alloc((long)FFD * EE * 2);
  bf16_t* woT   = (bf16_t*)alloc((long)EE * FFD * 2);
  bf16_t* qb    = (bf16_t*)alloc(T * EE * 2);
  bf16_t* kb    = (bf16_t*)alloc(T * EE * 2);
  bf16_t* vT    = (bf16_t*)alloc(T * EE * 2);
  float*  sc    = (float* )alloc((long)BB * SS * SS * 4);
  bf16_t* aw    = (bf16_t*)alloc((long)BB * SS * SS * 2);
  bf16_t* ao    = (bf16_t*)alloc(T * EE * 2);
  float*  h1pre = (float* )alloc(T * EE * 4);
  float*  h1f   = (float* )alloc(T * EE * 4);
  bf16_t* h1b   = (bf16_t*)alloc(T * EE * 2);
  bf16_t* ffb   = (bf16_t*)alloc(T * FFD * 2);
  float*  opre  = (float* )alloc(T * EE * 4);

  const dim3 tb(32, 8);

  // 1) fp32 -> bf16 conversions (weights transposed to [N][K])
  cvt_f32_bf16_kernel<<<2048, 256, 0, stream>>>(x, xb, T * EE);
  transpose_cvt_kernel<<<dim3(EE / 32, EE / 32), tb, 0, stream>>>(Wq, wqT, EE, EE);
  transpose_cvt_kernel<<<dim3(EE / 32, EE / 32), tb, 0, stream>>>(Wk, wkT, EE, EE);
  transpose_cvt_kernel<<<dim3(EE / 32, EE / 32), tb, 0, stream>>>(Wv, wvT, EE, EE);
  transpose_cvt_kernel<<<dim3(EE / 32, EE / 32), tb, 0, stream>>>(Wd, wdT, EE, EE);
  transpose_cvt_kernel<<<dim3(FFD / 32, EE / 32), tb, 0, stream>>>(Wi, wiT, EE, FFD);
  transpose_cvt_kernel<<<dim3(EE / 32, FFD / 32), tb, 0, stream>>>(Wo, woT, FFD, EE);

  // 2) q = x Wq + bq (bf16), k = x Wk + bk (bf16), vT = (x Wv + bv)^T per batch
  gemm_bf16_kernel<true, false, false, false, true, false>
      <<<dim3(EE / 128, T / 128, 1), 256, 0, stream>>>(
          xb, wqT, bq, nullptr, nullptr, qb, (int)T, EE, EE, 0, 0, 0, 0, 1, 1, 0);
  gemm_bf16_kernel<true, false, false, false, true, false>
      <<<dim3(EE / 128, T / 128, 1), 256, 0, stream>>>(
          xb, wkT, bk, nullptr, nullptr, kb, (int)T, EE, EE, 0, 0, 0, 0, 1, 1, 0);
  gemm_bf16_kernel<true, false, false, false, true, true>
      <<<dim3(EE / 128, T / 128, 1), 256, 0, stream>>>(
          xb, wvT, bv, nullptr, nullptr, vT, (int)T, EE, EE, 0, 0, 0, 0,
          SS, SS, (long)EE * SS);

  // 3) scores[b] = q[b] k[b]^T (fp32), batched over B
  gemm_bf16_kernel<false, false, false, true, false, false>
      <<<dim3(SS / 128, SS / 128, BB), 256, 0, stream>>>(
          qb, kb, nullptr, nullptr, sc, nullptr, SS, SS, EE,
          (long)SS * EE, (long)SS * EE, (long)SS * SS, 0, 1, 1, 0);

  // 4) attn_w = softmax(scores / 32) -> bf16
  softmax_bf16_kernel<<<BB * SS, 256, 0, stream>>>(sc, aw, 0.03125f);

  // 5) attn_out[b] = attn_w[b] V[b]  (Bt = vT[b] = V^T, [E][S])
  gemm_bf16_kernel<false, false, false, false, true, false>
      <<<dim3(EE / 128, SS / 128, BB), 256, 0, stream>>>(
          aw, vT, nullptr, nullptr, nullptr, ao, SS, EE, SS,
          (long)SS * SS, (long)EE * SS, 0, (long)SS * EE, 1, 1, 0);

  // 6) h1pre = attn_out Wd + bd + x (fp32)
  gemm_bf16_kernel<true, true, false, true, false, false>
      <<<dim3(EE / 128, T / 128, 1), 256, 0, stream>>>(
          ao, wdT, bd, x, h1pre, nullptr, (int)T, EE, EE, 0, 0, 0, 0, 1, 1, 0);

  // 7) h1 = LN(h1pre)*g1+b1 -> fp32 (residual) + bf16 (GEMM A)
  layernorm_kernel<true><<<(int)T, 256, 0, stream>>>(h1pre, g1, b1, h1f, h1b);

  // 8) ff = gelu(h1 Wi + bi) -> bf16
  gemm_bf16_kernel<true, false, true, false, true, false>
      <<<dim3(FFD / 128, T / 128, 1), 256, 0, stream>>>(
          h1b, wiT, bi, nullptr, nullptr, ffb, (int)T, FFD, EE, 0, 0, 0, 0, 1, 1, 0);

  // 9) opre = ff Wo + bo + h1 (fp32)
  gemm_bf16_kernel<true, true, false, true, false, false>
      <<<dim3(EE / 128, T / 128, 1), 256, 0, stream>>>(
          ffb, woT, bo, h1f, opre, nullptr, (int)T, EE, FFD, 0, 0, 0, 0, 1, 1, 0);

  // 10) out = LN(opre)*g2+b2 -> fp32 d_out
  layernorm_kernel<false><<<(int)T, 256, 0, stream>>>(opre, g2, b2, (float*)d_out, nullptr);
}
